// PNATransformerLayer_1082331758912
// MI455X (gfx1250) — matrix operand order repaired and checked by reading, not compiled, not run on hardware
//
#include <hip/hip_runtime.h>

typedef __attribute__((ext_vector_type(16))) __bf16    v16bf;
typedef __attribute__((ext_vector_type(8)))  float     v8f;
typedef __attribute__((ext_vector_type(4)))  unsigned  u32x4;
typedef unsigned short u16;

union Frag { v16bf v; u32x4 q[2]; };

__device__ __forceinline__ u16 f32_to_bf16(float f) {
  unsigned u = __float_as_uint(f);
  unsigned r = 0x7FFFu + ((u >> 16) & 1u);
  return (u16)((u + r) >> 16);
}
__device__ __forceinline__ unsigned fkey(float f) {
  unsigned u = __float_as_uint(f);
  return (u & 0x80000000u) ? ~u : (u | 0x80000000u);
}
__device__ __forceinline__ float funkey(unsigned k2) {
  unsigned u = (k2 & 0x80000000u) ? (k2 & 0x7FFFFFFFu) : ~k2;
  return __uint_as_float(u);
}
__device__ __forceinline__ v8f v8f_zero() {
  v8f z = {0.f,0.f,0.f,0.f,0.f,0.f,0.f,0.f}; return z;
}

// ---------------------------------------------------------------- utilities
__global__ void k_fill_u32(unsigned* __restrict__ p, unsigned val, long long n) {
  long long i = (long long)blockIdx.x * blockDim.x + threadIdx.x;
  if (i < n) p[i] = val;
}

__global__ void k_convert_bf16(const float* __restrict__ in, u16* __restrict__ out, long long n) {
  long long i = (long long)blockIdx.x * blockDim.x + threadIdx.x;
  if (i < n) out[i] = f32_to_bf16(in[i]);
}

// W f32 [K][Ncol] -> Wt bf16 [Ncol][K]
__global__ void k_transpose_bf16(const float* __restrict__ W, u16* __restrict__ Wt,
                                 int K, int Ncol) {
  long long i = (long long)blockIdx.x * blockDim.x + threadIdx.x;
  if (i >= (long long)K * Ncol) return;
  int kk = (int)(i / Ncol), n = (int)(i % Ncol);
  Wt[(long long)n * K + kk] = f32_to_bf16(W[i]);
}

// ---------------------------------------------------------------- PNA
__global__ void k_pna_deg(const int* __restrict__ dst, float* __restrict__ deg, long long E) {
  long long i = (long long)blockIdx.x * blockDim.x + threadIdx.x;
  if (i < E) atomicAdd(&deg[dst[i]], 1.0f);
}

__global__ void k_pna_edges(const float* __restrict__ nf, const int* __restrict__ src,
                            const int* __restrict__ dst, float* __restrict__ sum,
                            float* __restrict__ sumsq, unsigned* __restrict__ maxk,
                            unsigned* __restrict__ mink, long long total) {
  long long i = (long long)blockIdx.x * blockDim.x + threadIdx.x;
  if (i >= total) return;
  int e = (int)(i >> 6);
  int c = (int)(i & 63) * 4;
  int s = src[e], d = dst[e];
  const float* mrow = nf + (long long)s * 256 + c;
  long long ob = (long long)d * 256 + c;
#pragma unroll
  for (int j = 0; j < 4; ++j) {
    float m = mrow[j];
    atomicAdd(&sum[ob + j], m);
    atomicAdd(&sumsq[ob + j], m * m);
    atomicMax(&maxk[ob + j], fkey(m));
    atomicMin(&mink[ob + j], fkey(m));
  }
}

// emit scaled aggregate matrix [N, 8H] bf16: [mean,max,min,std, *log1p(deg)]
__global__ void k_pna_node_agg(const float* __restrict__ sum, const float* __restrict__ sumsq,
                               const unsigned* __restrict__ maxk, const unsigned* __restrict__ mink,
                               const float* __restrict__ deg, u16* __restrict__ agg8,
                               long long total) {
  long long i = (long long)blockIdx.x * blockDim.x + threadIdx.x;
  if (i >= total) return;
  long long n = i >> 8;
  int c = (int)(i & 255);
  float dg = deg[n];
  float safe = fmaxf(dg, 1.0f);
  long long idx = n * 256 + c;
  float mean = sum[idx] / safe;
  float msq  = sumsq[idx] / safe;
  float stdv = sqrtf(fmaxf(msq - mean * mean, 0.0f) + 1e-5f);
  bool has = dg > 0.0f;
  float mx = has ? funkey(maxk[idx]) : 0.0f;
  float mn = has ? funkey(mink[idx]) : 0.0f;
  float amp = log1pf(dg);
  u16* row = agg8 + n * 2048;
  row[c]        = f32_to_bf16(mean);
  row[256 + c]  = f32_to_bf16(mx);
  row[512 + c]  = f32_to_bf16(mn);
  row[768 + c]  = f32_to_bf16(stdv);
  row[1024 + c] = f32_to_bf16(mean * amp);
  row[1280 + c] = f32_to_bf16(mx * amp);
  row[1536 + c] = f32_to_bf16(mn * amp);
  row[1792 + c] = f32_to_bf16(stdv * amp);
}

// ---------------------------------------------------------------- WMMA GEMM
// C[M,N] = A[M,K](bf16) * Bt[N,K](bf16)^T  (+bias +resid, optional relu),
// out f32 and/or bf16. M,N mult of 128; K mult of 32.
#define GBM 128
#define GBN 128
#define GBK 32
#define GLDS 40   // padded halfword row stride (80B: bank-spread, 16B aligned)

__global__ __launch_bounds__(256) void k_gemm_bf16(
    const u16* __restrict__ A, const u16* __restrict__ Bt,
    float* __restrict__ Cf, u16* __restrict__ Cb,
    const float* __restrict__ bias, const float* __restrict__ resid,
    int M, int Nn, int K, int relu) {
  __shared__ __align__(16) u16 As[GBM * GLDS];
  __shared__ __align__(16) u16 Bs[GBN * GLDS];
  const int tid  = threadIdx.x;
  const int lane = tid & 31;
  const int wave = tid >> 5;
  const int wm = (wave & 3) * 32;
  const int wn = (wave >> 2) * 64;
  const long long m0 = (long long)blockIdx.y * GBM;
  const long long n0 = (long long)blockIdx.x * GBN;
  const int l15 = lane & 15;
  const int lhi = lane >> 4;
  const int lrow = tid >> 1;
  const int lseg = (tid & 1) * 16;

  v8f acc[2][4];
#pragma unroll
  for (int i = 0; i < 2; ++i)
#pragma unroll
    for (int j = 0; j < 4; ++j) acc[i][j] = v8f_zero();

  for (int k0 = 0; k0 < K; k0 += GBK) {
    const u16* gA = A + (m0 + lrow) * (long long)K + k0 + lseg;
    const u16* gB = Bt + (n0 + lrow) * (long long)K + k0 + lseg;
    u32x4 a0 = *(const u32x4*)gA;
    u32x4 a1 = *(const u32x4*)(gA + 8);
    u32x4 b0 = *(const u32x4*)gB;
    u32x4 b1 = *(const u32x4*)(gB + 8);
    __syncthreads();
    *(u32x4*)&As[lrow * GLDS + lseg]     = a0;
    *(u32x4*)&As[lrow * GLDS + lseg + 8] = a1;
    *(u32x4*)&Bs[lrow * GLDS + lseg]     = b0;
    *(u32x4*)&Bs[lrow * GLDS + lseg + 8] = b1;
    __syncthreads();

    Frag af[2];
#pragma unroll
    for (int mi = 0; mi < 2; ++mi) {
      const u16* ap = &As[(wm + mi * 16 + l15) * GLDS + (lhi << 3)];
      af[mi].q[0] = *(const u32x4*)ap;        // K = baseK .. baseK+7
      af[mi].q[1] = *(const u32x4*)(ap + 16); // K = baseK+16 .. +23
    }
#pragma unroll
    for (int ni = 0; ni < 4; ++ni) {
      Frag bf;
      const u16* bp = &Bs[(wn + ni * 16 + l15) * GLDS + (lhi << 3)];
      bf.q[0] = *(const u32x4*)bp;            // K = baseK .. baseK+7
      bf.q[1] = *(const u32x4*)(bp + 16);     // K = baseK+16 .. +23
#pragma unroll
      for (int mi = 0; mi < 2; ++mi)
        acc[mi][ni] = __builtin_amdgcn_wmma_f32_16x16x32_bf16(
            false, af[mi].v, false, bf.v, (short)0, acc[mi][ni], false, false);
    }
  }

#pragma unroll
  for (int mi = 0; mi < 2; ++mi)
#pragma unroll
    for (int ni = 0; ni < 4; ++ni)
#pragma unroll
      for (int r = 0; r < 8; ++r) {
        long long row = m0 + wm + mi * 16 + r + (lhi << 3);
        long long col = n0 + wn + ni * 16 + l15;
        float vv = acc[mi][ni][r];
        if (bias)  vv += bias[col];
        if (resid) vv += resid[row * Nn + col];
        if (relu)  vv = fmaxf(vv, 0.0f);
        if (Cf) Cf[row * Nn + col] = vv;
        if (Cb) Cb[row * Nn + col] = f32_to_bf16(vv);
      }
}

// ---------------------------------------------------------------- attention
// grid = B*NH; block 256 (8 waves, 16 query rows each). S=128, HD=64, H=256.
__global__ __launch_bounds__(256) void k_attention(
    const u16* __restrict__ q, const u16* __restrict__ k,
    const u16* __restrict__ v, u16* __restrict__ o) {
  __shared__ __align__(16) u16 Vt[64 * 136];      // V^T: [d][s]
  __shared__ __align__(16) u16 Ps[8 * 16 * 136];  // per-wave P tile [16][s]
  const int tid = threadIdx.x;
  const int lane = tid & 31;
  const int wave = tid >> 5;
  const int l15 = lane & 15;
  const int lhi = lane >> 4;
  const int b = blockIdx.x >> 2;
  const int head = blockIdx.x & 3;
  const long long base = ((long long)b * 128) * 256 + head * 64;

  { // transpose V into LDS (coalesced global reads, scattered ds stores)
    int row = tid >> 1;
    int seg = (tid & 1) * 32;
    const u16* src = v + base + (long long)row * 256 + seg;
    u16 tmp[32];
#pragma unroll
    for (int j = 0; j < 4; ++j) *(u32x4*)&tmp[j * 8] = *(const u32x4*)(src + j * 8);
#pragma unroll
    for (int j = 0; j < 32; ++j) Vt[(seg + j) * 136 + row] = tmp[j];
  }
  __syncthreads();

  const int s0 = wave * 16;
  Frag qf[2];
  {
    const u16* qp = q + base + (long long)(s0 + l15) * 256 + (lhi << 3);
#pragma unroll
    for (int ks = 0; ks < 2; ++ks) {
      qf[ks].q[0] = *(const u32x4*)(qp + ks * 32);
      qf[ks].q[1] = *(const u32x4*)(qp + ks * 32 + 16);
    }
  }

  // scores = Q * K^T (B-frags straight from global: contiguous in d)
  v8f sacc[8];
#pragma unroll
  for (int nt = 0; nt < 8; ++nt) sacc[nt] = v8f_zero();
#pragma unroll
  for (int ks = 0; ks < 2; ++ks)
#pragma unroll
    for (int nt = 0; nt < 8; ++nt) {
      Frag bfr;
      const u16* kp = k + base + (long long)(nt * 16 + l15) * 256 + ks * 32 + (lhi << 3);
      bfr.q[0] = *(const u32x4*)kp;
      bfr.q[1] = *(const u32x4*)(kp + 16);
      sacc[nt] = __builtin_amdgcn_wmma_f32_16x16x32_bf16(
          false, qf[ks].v, false, bfr.v, (short)0, sacc[nt], false, false);
    }

  // scale, mask (key pos > 100), softmax over 128 keys per row
  u16* Pw = &Ps[wave * 16 * 136];
#pragma unroll
  for (int r = 0; r < 8; ++r) {
    float mx = -1e30f;
#pragma unroll
    for (int nt = 0; nt < 8; ++nt) {
      float sv = sacc[nt][r] * 0.125f;           // 1/sqrt(64)
      if (nt * 16 + l15 > 100) sv = -1e9f;
      sacc[nt][r] = sv;
      mx = fmaxf(mx, sv);
    }
#pragma unroll
    for (int off = 1; off < 16; off <<= 1) mx = fmaxf(mx, __shfl_xor(mx, off, 32));
    float sm = 0.0f;
#pragma unroll
    for (int nt = 0; nt < 8; ++nt) {
      float e = __expf(sacc[nt][r] - mx);
      sacc[nt][r] = e;
      sm += e;
    }
#pragma unroll
    for (int off = 1; off < 16; off <<= 1) sm += __shfl_xor(sm, off, 32);
    float inv = 1.0f / sm;
#pragma unroll
    for (int nt = 0; nt < 8; ++nt)
      Pw[(r + (lhi << 3)) * 136 + nt * 16 + l15] = f32_to_bf16(sacc[nt][r] * inv);
  }
  __syncthreads();

  // O = P * V
  v8f oacc[4];
#pragma unroll
  for (int nt = 0; nt < 4; ++nt) oacc[nt] = v8f_zero();
#pragma unroll
  for (int ks = 0; ks < 4; ++ks) {
    Frag pf;
    const u16* pp = &Pw[l15 * 136 + ks * 32 + (lhi << 3)];
    pf.q[0] = *(const u32x4*)pp;
    pf.q[1] = *(const u32x4*)(pp + 16);
#pragma unroll
    for (int nt = 0; nt < 4; ++nt) {
      Frag vf;
      const u16* vp = &Vt[(nt * 16 + l15) * 136 + ks * 32 + (lhi << 3)];
      vf.q[0] = *(const u32x4*)vp;
      vf.q[1] = *(const u32x4*)(vp + 16);
      oacc[nt] = __builtin_amdgcn_wmma_f32_16x16x32_bf16(
          false, pf.v, false, vf.v, (short)0, oacc[nt], false, false);
    }
  }
#pragma unroll
  for (int nt = 0; nt < 4; ++nt)
#pragma unroll
    for (int r = 0; r < 8; ++r) {
      int row = s0 + r + (lhi << 3);
      int col = nt * 16 + l15;
      o[base + (long long)row * 256 + col] = f32_to_bf16(oacc[nt][r]);
    }
}

// ---------------------------------------------------------------- layernorm
__global__ __launch_bounds__(256) void k_layernorm(
    const float* __restrict__ in, const float* __restrict__ g, const float* __restrict__ bb,
    float* __restrict__ outf, u16* __restrict__ outb) {
  __shared__ float red[256];
  int row = blockIdx.x, t = threadIdx.x;
  float x = in[(long long)row * 256 + t];
  red[t] = x; __syncthreads();
  for (int s = 128; s > 0; s >>= 1) { if (t < s) red[t] += red[t + s]; __syncthreads(); }
  float mu = red[0] * (1.0f / 256.0f);
  __syncthreads();
  float d = x - mu;
  red[t] = d * d; __syncthreads();
  for (int s = 128; s > 0; s >>= 1) { if (t < s) red[t] += red[t + s]; __syncthreads(); }
  float var = red[0] * (1.0f / 256.0f);
  float y = d * rsqrtf(var + 1e-5f) * g[t] + bb[t];
  if (outf) outf[(long long)row * 256 + t] = y;
  if (outb) outb[(long long)row * 256 + t] = f32_to_bf16(y);
}

// ---------------------------------------------------------------- combine pack
__global__ void k_pack_combine(const float* __restrict__ h, const u16* __restrict__ hgb,
                               const u16* __restrict__ htb, u16* __restrict__ ac,
                               long long total) {
  long long i = (long long)blockIdx.x * blockDim.x + threadIdx.x;
  if (i >= total) return;
  long long row = i >> 9;
  int c = (int)(i & 511);
  u16 val;
  if (c < 256) {
    int s = (int)(row & 127);
    if (s >= 1 && s <= 100) {
      long long b = row >> 7;
      val = hgb[(b * 100 + (s - 1)) * 256 + c];
    } else {
      val = f32_to_bf16(h[row * 256 + c]);
    }
  } else {
    val = htb[row * 256 + (c - 256)];
  }
  ac[i] = val;
}

// ---------------------------------------------------------------- host
static inline unsigned cdivll(long long a, long long b) { return (unsigned)((a + b - 1) / b); }

extern "C" void kernel_launch(void* const* d_in, const int* in_sizes, int n_in,
                              void* d_out, int out_size, void* d_ws, size_t ws_size,
                              hipStream_t stream) {
  const float* h   = (const float*)d_in[0];
  const float* nf  = (const float*)d_in[1];
  const int*   src = (const int*)d_in[2];
  const int*   dst = (const int*)d_in[3];
  const float* Wp  = (const float*)d_in[6];
  const float* bp  = (const float*)d_in[7];
  const float* Wq  = (const float*)d_in[8];
  const float* Wk  = (const float*)d_in[9];
  const float* Wv  = (const float*)d_in[10];
  const float* Wo  = (const float*)d_in[11];
  const float* bq  = (const float*)d_in[12];
  const float* bk  = (const float*)d_in[13];
  const float* bv  = (const float*)d_in[14];
  const float* bo  = (const float*)d_in[15];
  const float* g1  = (const float*)d_in[16];
  const float* be1 = (const float*)d_in[17];
  const float* W1  = (const float*)d_in[18];
  const float* b1  = (const float*)d_in[19];
  const float* W2  = (const float*)d_in[20];
  const float* b2  = (const float*)d_in[21];
  const float* g2  = (const float*)d_in[22];
  const float* be2 = (const float*)d_in[23];
  const float* Wc  = (const float*)d_in[24];
  const float* bc  = (const float*)d_in[25];
  float* out = (float*)d_out;
  (void)in_sizes; (void)n_in; (void)out_size; (void)ws_size;

  const long long Nn = 51200, Ee = 409600, BS = 65536;
  const int H = 256, FF = 1024;

  char* ws = (char*)d_ws;
  size_t off = 0;
  auto alloc = [&](size_t bytes) { size_t p = off; off += (bytes + 255) & ~(size_t)255; return p; };

  // Phase-1 (PNA) scratch; region reused by phase-2 activations afterwards.
  size_t o_deg   = alloc((size_t)Nn * 4);
  size_t o_sum   = alloc((size_t)Nn * H * 4);
  size_t o_sumsq = alloc((size_t)Nn * H * 4);
  size_t o_maxk  = alloc((size_t)Nn * H * 4);
  size_t o_mink  = alloc((size_t)Nn * H * 4);
  size_t o_agg8  = alloc((size_t)Nn * 2048 * 2);
  // Persistent
  size_t o_hgb = alloc((size_t)Nn * H * 2);
  size_t o_Wpt = alloc((size_t)2048 * 256 * 2);
  size_t o_Wqt = alloc((size_t)256 * 256 * 2);
  size_t o_Wkt = alloc((size_t)256 * 256 * 2);
  size_t o_Wvt = alloc((size_t)256 * 256 * 2);
  size_t o_Wot = alloc((size_t)256 * 256 * 2);
  size_t o_W1t = alloc((size_t)256 * 1024 * 2);
  size_t o_W2t = alloc((size_t)1024 * 256 * 2);
  size_t o_Wct = alloc((size_t)512 * 256 * 2);
  size_t o_t1  = alloc((size_t)BS * H * 4);   // reused as t2
  size_t o_x   = alloc((size_t)BS * H * 4);   // reused as Acomb (same byte size)
  // Phase-2 overlays of PNA scratch (dead after posttrans GEMM)
  size_t o_q  = o_deg;
  size_t o_k  = o_q + (size_t)BS * H * 2;
  size_t o_v  = o_k + (size_t)BS * H * 2;
  size_t o_ob = o_v + (size_t)BS * H * 2;
  size_t o_xb = o_ob + (size_t)BS * H * 2;
  size_t o_ff1 = o_agg8;
  size_t o_hb  = o_ff1 + (size_t)BS * FF * 2;
  size_t o_htb = o_hb + (size_t)BS * H * 2;

  float*    deg   = (float*)(ws + o_deg);
  float*    sum   = (float*)(ws + o_sum);
  float*    sumsq = (float*)(ws + o_sumsq);
  unsigned* maxk  = (unsigned*)(ws + o_maxk);
  unsigned* mink  = (unsigned*)(ws + o_mink);
  u16* agg8 = (u16*)(ws + o_agg8);
  u16* hgb  = (u16*)(ws + o_hgb);
  u16* Wpt  = (u16*)(ws + o_Wpt);
  u16* Wqt  = (u16*)(ws + o_Wqt);
  u16* Wkt  = (u16*)(ws + o_Wkt);
  u16* Wvt  = (u16*)(ws + o_Wvt);
  u16* Wot  = (u16*)(ws + o_Wot);
  u16* W1t  = (u16*)(ws + o_W1t);
  u16* W2t  = (u16*)(ws + o_W2t);
  u16* Wct  = (u16*)(ws + o_Wct);
  float* t1 = (float*)(ws + o_t1);
  float* t2 = (float*)(ws + o_t1);
  float* x  = (float*)(ws + o_x);
  u16* acomb = (u16*)(ws + o_x);
  u16* qb  = (u16*)(ws + o_q);
  u16* kb  = (u16*)(ws + o_k);
  u16* vb  = (u16*)(ws + o_v);
  u16* ob  = (u16*)(ws + o_ob);
  u16* xb  = (u16*)(ws + o_xb);
  u16* ff1 = (u16*)(ws + o_ff1);
  u16* hb  = (u16*)(ws + o_hb);
  u16* htb = (u16*)(ws + o_htb);

  dim3 blk(256);

  // ---- PNA ----
  long long zn = (long long)(o_mink - o_deg) / 4;
  k_fill_u32<<<cdivll(zn, 256), blk, 0, stream>>>((unsigned*)(ws + o_deg), 0u, zn);
  k_fill_u32<<<cdivll(Nn * H, 256), blk, 0, stream>>>(mink, 0xFFFFFFFFu, Nn * H);
  k_pna_deg<<<cdivll(Ee, 256), blk, 0, stream>>>(dst, deg, Ee);
  k_pna_edges<<<cdivll(Ee * 64, 256), blk, 0, stream>>>(nf, src, dst, sum, sumsq, maxk, mink, Ee * 64);
  k_pna_node_agg<<<cdivll(Nn * H, 256), blk, 0, stream>>>(sum, sumsq, maxk, mink, deg, agg8, Nn * H);
  k_transpose_bf16<<<cdivll(2048LL * 256, 256), blk, 0, stream>>>(Wp, Wpt, 2048, 256);
  dim3 gPost(2, (unsigned)(Nn / 128));
  k_gemm_bf16<<<gPost, blk, 0, stream>>>(agg8, Wpt, nullptr, hgb, bp, nf, (int)Nn, 256, 2048, 0);

  // ---- weight prep + h->bf16 ----
  k_transpose_bf16<<<cdivll(65536, 256), blk, 0, stream>>>(Wq, Wqt, 256, 256);
  k_transpose_bf16<<<cdivll(65536, 256), blk, 0, stream>>>(Wk, Wkt, 256, 256);
  k_transpose_bf16<<<cdivll(65536, 256), blk, 0, stream>>>(Wv, Wvt, 256, 256);
  k_transpose_bf16<<<cdivll(65536, 256), blk, 0, stream>>>(Wo, Wot, 256, 256);
  k_transpose_bf16<<<cdivll(262144, 256), blk, 0, stream>>>(W1, W1t, 256, 1024);
  k_transpose_bf16<<<cdivll(262144, 256), blk, 0, stream>>>(W2, W2t, 1024, 256);
  k_transpose_bf16<<<cdivll(131072, 256), blk, 0, stream>>>(Wc, Wct, 512, 256);
  k_convert_bf16<<<cdivll(BS * H, 256), blk, 0, stream>>>(h, hb, BS * (long long)H);

  // ---- transformer ----
  dim3 gQKV(2, (unsigned)(BS / 128));
  k_gemm_bf16<<<gQKV, blk, 0, stream>>>(hb, Wqt, nullptr, qb, bq, nullptr, (int)BS, 256, 256, 0);
  k_gemm_bf16<<<gQKV, blk, 0, stream>>>(hb, Wkt, nullptr, kb, bk, nullptr, (int)BS, 256, 256, 0);
  k_gemm_bf16<<<gQKV, blk, 0, stream>>>(hb, Wvt, nullptr, vb, bv, nullptr, (int)BS, 256, 256, 0);
  k_attention<<<512 * 4, blk, 0, stream>>>(qb, kb, vb, ob);
  k_gemm_bf16<<<gQKV, blk, 0, stream>>>(ob, Wot, t1, nullptr, bo, h, (int)BS, 256, 256, 0);
  k_layernorm<<<(unsigned)BS, blk, 0, stream>>>(t1, g1, be1, x, xb);
  dim3 gFF1(8, (unsigned)(BS / 128));
  k_gemm_bf16<<<gFF1, blk, 0, stream>>>(xb, W1t, nullptr, ff1, b1, nullptr, (int)BS, FF, 256, 1);
  k_gemm_bf16<<<gQKV, blk, 0, stream>>>(ff1, W2t, t2, nullptr, b2, x, (int)BS, 256, FF, 0);
  k_layernorm<<<(unsigned)BS, blk, 0, stream>>>(t2, g2, be2, nullptr, htb);

  // ---- combine ----
  k_pack_combine<<<cdivll(BS * 512, 256), blk, 0, stream>>>(h, hgb, htb, acomb, BS * 512LL);
  k_gemm_bf16<<<gQKV, blk, 0, stream>>>(acomb, Wct, out, nullptr, bc, nullptr, (int)BS, 256, 512, 0);
}
